// callPredictor_26396869001742
// MI455X (gfx1250) — matrix operand order repaired and checked by reading, not compiled, not run on hardware
//
#include <hip/hip_runtime.h>

// Problem constants (from reference)
#define B_   256
#define TS_  512
#define F_   256
#define W_   3
#define H_   256
#define G4_  1024
#define ROWS 16          // batch rows (WMMA M tile) per workgroup
#define NTHREADS 256     // 8 waves (wave32)

typedef __attribute__((ext_vector_type(16))) _Float16 v16h;
typedef __attribute__((ext_vector_type(8)))  _Float16 v8h;
typedef __attribute__((ext_vector_type(8)))  float    v8f;

// f16 weight workspace layout (element offsets):
//   WhbT : F_  x H_  at 0        (WhbT[n*H_+k] = Whb[k][n])
//   WihT : G4_ x F_  at 65536
//   WhhT : G4_ x H_  at 327680
#define OFF_WHB 0
#define OFF_WIH 65536
#define OFF_WHH (65536 + 262144)
#define CONV_TOTAL (65536 + 262144 + 262144)

__global__ void convert_weights(const float* __restrict__ Whb,
                                const float* __restrict__ Wih,
                                const float* __restrict__ Whh,
                                _Float16* __restrict__ ws) {
  for (int i = blockIdx.x * blockDim.x + threadIdx.x; i < CONV_TOTAL;
       i += gridDim.x * blockDim.x) {
    if (i < OFF_WIH) {
      int n = i >> 8, k = i & 255;
      ws[i] = (_Float16)Whb[k * F_ + n];
    } else if (i < OFF_WHH) {
      int j = i - OFF_WIH;
      int n = j >> 8, k = j & 255;
      ws[i] = (_Float16)Wih[k * G4_ + n];
    } else {
      int j = i - OFF_WHH;
      int n = j >> 8, k = j & 255;
      ws[i] = (_Float16)Whh[k * G4_ + n];
    }
  }
}

__device__ __forceinline__ float sigmoidf_(float x) {
  return 1.0f / (1.0f + __expf(-x));
}

// Async DMA: copy 16 bytes global -> LDS, tracked by ASYNCcnt (CDNA5).
__device__ __forceinline__ void async_g2l_b128(unsigned lds_off,
                                               const void* gaddr) {
  asm volatile("global_load_async_to_lds_b128 %0, %1, off"
               :
               : "v"(lds_off), "v"((unsigned long long)(uintptr_t)gaddr)
               : "memory");
}
__device__ __forceinline__ void wait_asynccnt0() {
  asm volatile("s_wait_asynccnt 0x0" ::: "memory");
}

// A fragment (16x32 f16, M=row stripe): lanes 0-15 -> M, contiguous K per half:
// elems 0-7 = K kb+kh+0..7, elems 8-15 = K kb+kh+8..15 (kh = 0 or 16)
__device__ __forceinline__ v16h load_a_frag(const _Float16* Ash, int lane, int kb) {
  int m  = lane & 15;
  int kh = (lane >> 4) << 4;  // 0 or 16
  const _Float16* row = Ash + m * 256;
  v8h lo = *reinterpret_cast<const v8h*>(row + kb + kh);
  v8h hi = *reinterpret_cast<const v8h*>(row + kb + kh + 8);
  return __builtin_shufflevector(lo, hi, 0, 1, 2, 3, 4, 5, 6, 7,
                                 8, 9, 10, 11, 12, 13, 14, 15);
}

// B fragment (32x16 f16, col-per-lane): lane n holds K = ks..ks+15 of col n.
// Weights stored N-major (WT[n*256 + k]) so this is one contiguous 32B load.
__device__ __forceinline__ v16h load_b_frag(const _Float16* WT, int lane,
                                            int ntile, int kb) {
  int n  = lane & 15;
  int ks = (lane >> 4) << 4;  // 0 or 16
  return *reinterpret_cast<const v16h*>(WT + (size_t)(ntile * 16 + n) * 256 + kb + ks);
}

#define WMMA_F16(a, b, c)                                                  \
  __builtin_amdgcn_wmma_f32_16x16x32_f16(false, (a), false, (b), (short)0, \
                                         (c), false, false)

__global__ void __launch_bounds__(NTHREADS)
attn_lstm_kernel(const float* __restrict__ X, const float* __restrict__ M,
                 const float* __restrict__ curM,
                 const float* __restrict__ Wha, const float* __restrict__ bha,
                 const float* __restrict__ Waa, const float* __restrict__ baa,
                 const float* __restrict__ bhb,
                 const float* __restrict__ Wab, const float* __restrict__ bab,
                 const float* __restrict__ bih, const float* __restrict__ bhh,
                 const float* __restrict__ Wp, const float* __restrict__ bp,
                 const _Float16* __restrict__ ws16, float* __restrict__ out) {
  __shared__ __align__(16) float    xw[ROWS][W_][F_];     // 48 KB window
  __shared__ __align__(16) float    cst[ROWS][H_];        // c state (f32)
  __shared__ __align__(16) _Float16 h16buf[2][ROWS][H_];  // ping-pong h (f16)
  __shared__ __align__(16) _Float16 z16[ROWS][F_];        // z (f16, WMMA A)
  __shared__ __align__(16) float    beta[ROWS][F_];
  __shared__ float apart[ROWS][16][W_];
  __shared__ float alpha[ROWS][4];
  __shared__ float part16[ROWS][16];
  __shared__ float rowred[ROWS];
  __shared__ float curacc[ROWS];

  const int tid  = threadIdx.x;
  const int lane = tid & 31;
  const int wv   = tid >> 5;   // wave id 0..7
  const int blk  = blockIdx.x;
  const int r16  = tid >> 4;   // row 0..15
  const int p16  = tid & 15;   // partial slot 0..15

  const _Float16* WhbT = ws16 + OFF_WHB;
  const _Float16* WihT = ws16 + OFF_WIH;
  const _Float16* WhhT = ws16 + OFF_WHH;

  const unsigned xw_lds_base = (unsigned)(uintptr_t)&xw[0][0][0];

  for (int i = tid; i < ROWS * H_; i += NTHREADS) {
    cst[i >> 8][i & 255] = 0.0f;
    h16buf[0][i >> 8][i & 255] = (_Float16)0.0f;
  }
  if (tid < ROWS) curacc[tid] = 0.0f;
  __syncthreads();

  for (int t = 0; t < TS_; ++t) {
    const _Float16* hc = &h16buf[t & 1][0][0];        // h_t (read)
    _Float16*       hn = &h16buf[(t + 1) & 1][0][0];  // h_{t+1} (write)

    // ---- 1) launch async DMA of the x window into LDS (ASYNCcnt) ----
    for (int q = tid; q < ROWS * W_ * (F_ / 4); q += NTHREADS) {
      int r    = q / (W_ * (F_ / 4));
      int rem  = q - r * (W_ * (F_ / 4));
      int w    = rem / (F_ / 4);
      int f4   = rem - w * (F_ / 4);
      int tsrc = t - (W_ - 1) + w;
      unsigned ldsoff =
          xw_lds_base + (unsigned)(((r * W_ + w) * F_ + f4 * 4) * sizeof(float));
      if (tsrc >= 0) {
        const float* gsrc =
            X + ((size_t)(blk * ROWS + r) * TS_ + tsrc) * F_ + f4 * 4;
        async_g2l_b128(ldsoff, gsrc);
      } else {
        reinterpret_cast<float4*>(&xw[r][w][0])[f4] =
            make_float4(0.f, 0.f, 0.f, 0.f);
      }
    }

    // ---- 2) beta pre-act: h @ Whb + bhb via WMMA, overlapped with the DMA.
    // Two independent accumulator chains per wave; fragments prefetched one
    // k-step ahead; unroll-2 turns the pipeline rotation into register
    // renaming (no v_mov / hazard nops).
    {
      int nt0 = 2 * wv, nt1 = 2 * wv + 1;
      float b0 = bhb[nt0 * 16 + (lane & 15)];
      float b1 = bhb[nt1 * 16 + (lane & 15)];
      v8f acc0, acc1;
#pragma unroll
      for (int r = 0; r < 8; ++r) { acc0[r] = b0; acc1[r] = b1; }
      v16h a  = load_a_frag(hc, lane, 0);
      v16h w0 = load_b_frag(WhbT, lane, nt0, 0);
      v16h w1 = load_b_frag(WhbT, lane, nt1, 0);
#pragma unroll 2
      for (int kb = 0; kb < H_; kb += 32) {
        int kn = (kb + 32) & (H_ - 1);
        v16h an  = load_a_frag(hc, lane, kn);
        v16h w0n = load_b_frag(WhbT, lane, nt0, kn);
        v16h w1n = load_b_frag(WhbT, lane, nt1, kn);
        acc0 = WMMA_F16(a, w0, acc0);
        acc1 = WMMA_F16(a, w1, acc1);
        a = an; w0 = w0n; w1 = w1n;
      }
      int m0 = (lane >> 4) << 3;
      int n0 = nt0 * 16 + (lane & 15);
      int n1 = nt1 * 16 + (lane & 15);
#pragma unroll
      for (int r = 0; r < 8; ++r) {
        beta[m0 + r][n0] = acc0[r];
        beta[m0 + r][n1] = acc1[r];
      }
    }
    wait_asynccnt0();   // xw DMA complete (this wave's ops)
    __syncthreads();    // beta + xw visible to all

    // ---- 3) alpha partial dots; beta finalize (tanh of pre + Wab term) ----
    {
      float a0 = 0.f, a1 = 0.f, a2 = 0.f;
#pragma unroll 4
      for (int j = 0; j < 16; ++j) {
        int k = p16 * 16 + j;
        float hv = (float)hc[r16 * H_ + k];
        float xa = Waa[k];
        a0 += hv * Wha[k * 3 + 0] + xw[r16][0][k] * xa;
        a1 += hv * Wha[k * 3 + 1] + xw[r16][1][k] * xa;
        a2 += hv * Wha[k * 3 + 2] + xw[r16][2][k] * xa;
      }
      apart[r16][p16][0] = a0;
      apart[r16][p16][1] = a1;
      apart[r16][p16][2] = a2;
    }
    {
      float w0 = Wab[0], w1 = Wab[1], w2 = Wab[2], bb = bab[0];
      int fb = p16 * 16;
#pragma unroll 4
      for (int j = 0; j < 16; ++j) {
        int f   = fb + j;
        float v = beta[r16][f] + xw[r16][0][f] * w0 + xw[r16][1][f] * w1 +
                  xw[r16][2][f] * w2 + bb;
        beta[r16][f] = tanhf(v);
      }
    }
    __syncthreads();

    // ---- 4) alpha reduce + tanh; beta softmax row-max partials ----
    if (tid < ROWS * W_) {
      int r = tid / W_, w = tid - (tid / W_) * W_;
      float s = 0.f;
#pragma unroll
      for (int p = 0; p < 16; ++p) s += apart[r][p][w];
      alpha[r][w] = tanhf(s + bha[w] + baa[0]);
    }
    {
      int fb = p16 * 16;
      float mx = beta[r16][fb];
#pragma unroll
      for (int j = 1; j < 16; ++j) mx = fmaxf(mx, beta[r16][fb + j]);
      part16[r16][p16] = mx;
    }
    __syncthreads();

    // ---- 5) alpha softmax (width 3); beta row max reduce ----
    if (tid < ROWS) {
      float a0 = alpha[tid][0], a1 = alpha[tid][1], a2 = alpha[tid][2];
      float mx = fmaxf(a0, fmaxf(a1, a2));
      float e0 = __expf(a0 - mx), e1 = __expf(a1 - mx), e2 = __expf(a2 - mx);
      float inv = 1.f / (e0 + e1 + e2);
      alpha[tid][0] = e0 * inv;
      alpha[tid][1] = e1 * inv;
      alpha[tid][2] = e2 * inv;
      float bmx = part16[tid][0];
#pragma unroll
      for (int p = 1; p < 16; ++p) bmx = fmaxf(bmx, part16[tid][p]);
      rowred[tid] = bmx;
    }
    __syncthreads();
    {
      float mx = rowred[r16];
      int fb = p16 * 16;
      float s = 0.f;
#pragma unroll 4
      for (int j = 0; j < 16; ++j) {
        int f = fb + j;
        float e = __expf(beta[r16][f] - mx);
        beta[r16][f] = e;
        s += e;
      }
      part16[r16][p16] = s;
    }
    __syncthreads();
    if (tid < ROWS) {
      float s = 0.f;
#pragma unroll
      for (int p = 0; p < 16; ++p) s += part16[tid][p];
      rowred[tid] = 1.f / s;
    }
    __syncthreads();

    // ---- 6) z = softmax(beta) * (alpha . xw), packed to f16 ----
    {
      float inv = rowred[r16];
      float al0 = alpha[r16][0], al1 = alpha[r16][1], al2 = alpha[r16][2];
      int fb = p16 * 16;
#pragma unroll 4
      for (int j = 0; j < 16; ++j) {
        int f = fb + j;
        float bv = beta[r16][f] * inv;
        float xv = al0 * xw[r16][0][f] + al1 * xw[r16][1][f] + al2 * xw[r16][2][f];
        z16[r16][f] = (_Float16)(bv * xv);
      }
    }
    __syncthreads();

    // ---- 7) gates = z@Wih + h@Whh + biases; LSTM elementwise ----
    // Wave wv owns hidden columns [wv*32, wv*32+32): two 16-col passes.
    // 4 independent accumulator chains per pass hide WMMA RAW latency;
    // A-frags and the g=0 B-frags are prefetched one k-step ahead.
#pragma unroll 1
    for (int ntl = 0; ntl < 2; ++ntl) {
      int hc16 = wv * 32 + ntl * 16;
      int tbase = wv * 2 + ntl;
      v8f acc[4];
#pragma unroll
      for (int g = 0; g < 4; ++g) {
        int col    = g * 256 + hc16 + (lane & 15);
        float bias = bih[col] + bhh[col];
#pragma unroll
        for (int r = 0; r < 8; ++r) acc[g][r] = bias;
      }
      v16h az  = load_a_frag(&z16[0][0], lane, 0);
      v16h ah  = load_a_frag(hc, lane, 0);
      v16h bz0 = load_b_frag(WihT, lane, tbase, 0);
      v16h bh0 = load_b_frag(WhhT, lane, tbase, 0);
#pragma unroll 2
      for (int kb = 0; kb < 256; kb += 32) {
        int kn = (kb + 32) & 255;
        v16h azn  = load_a_frag(&z16[0][0], lane, kn);
        v16h ahn  = load_a_frag(hc, lane, kn);
        v16h bz0n = load_b_frag(WihT, lane, tbase, kn);
        v16h bh0n = load_b_frag(WhhT, lane, tbase, kn);
        acc[0] = WMMA_F16(az, bz0, acc[0]);
        acc[0] = WMMA_F16(ah, bh0, acc[0]);
#pragma unroll
        for (int g = 1; g < 4; ++g) {
          int ntile = g * 16 + tbase;
          v16h bz = load_b_frag(WihT, lane, ntile, kb);
          acc[g] = WMMA_F16(az, bz, acc[g]);
          v16h bh = load_b_frag(WhhT, lane, ntile, kb);
          acc[g] = WMMA_F16(ah, bh, acc[g]);
        }
        az = azn; ah = ahn; bz0 = bz0n; bh0 = bh0n;
      }
      int hcol = hc16 + (lane & 15);
      int m0   = (lane >> 4) << 3;
#pragma unroll
      for (int r = 0; r < 8; ++r) {
        int m = m0 + r;
        float iv = sigmoidf_(acc[0][r]);
        float fv = sigmoidf_(acc[1][r]);
        float gv = tanhf(acc[2][r]);
        float ov = sigmoidf_(acc[3][r]);
        float cn = fv * cst[m][hcol] + iv * gv;
        float hnv = ov * tanhf(cn);
        cst[m][hcol] = cn;
        hn[m * H_ + hcol] = (_Float16)hnv;
      }
    }
    __syncthreads();

    // ---- 8) output projection: (h_new . Wp + bp) * M; accumulate cur ----
    {
      float s = 0.f;
#pragma unroll 4
      for (int j = 0; j < 16; ++j) {
        int k = p16 * 16 + j;
        s += (float)hn[r16 * H_ + k] * Wp[k];
      }
      part16[r16][p16] = s;
    }
    __syncthreads();
    if (tid < ROWS) {
      float s = 0.f;
#pragma unroll
      for (int p = 0; p < 16; ++p) s += part16[tid][p];
      float o   = s + bp[0];
      size_t b  = (size_t)blk * ROWS + tid;
      float val = o * M[b * TS_ + t];
      out[b * TS_ + t] = val;
      curacc[tid] += val * curM[b * TS_ + t];
    }
    __syncthreads();
  }

  if (tid < ROWS) {
    size_t b = (size_t)blk * ROWS + tid;
    out[(size_t)B_ * TS_ + b] = curacc[tid];
  }
}

extern "C" void kernel_launch(void* const* d_in, const int* in_sizes, int n_in,
                              void* d_out, int out_size, void* d_ws,
                              size_t ws_size, hipStream_t stream) {
  const float* X    = (const float*)d_in[0];
  const float* M    = (const float*)d_in[1];
  const float* curM = (const float*)d_in[2];
  // d_in[3] = T (int32), unused by the reference math
  const float* Wha  = (const float*)d_in[4];
  const float* bha  = (const float*)d_in[5];
  const float* Waa  = (const float*)d_in[6];
  const float* baa  = (const float*)d_in[7];
  const float* Whb  = (const float*)d_in[8];
  const float* bhb  = (const float*)d_in[9];
  const float* Wab  = (const float*)d_in[10];
  const float* bab  = (const float*)d_in[11];
  const float* Wih  = (const float*)d_in[12];
  const float* Whh  = (const float*)d_in[13];
  const float* bih  = (const float*)d_in[14];
  const float* bhh  = (const float*)d_in[15];
  const float* Wp   = (const float*)d_in[16];
  const float* bp   = (const float*)d_in[17];
  float* out        = (float*)d_out;
  _Float16* ws16    = (_Float16*)d_ws;

  hipLaunchKernelGGL(convert_weights, dim3(512), dim3(256), 0, stream,
                     Whb, Wih, Whh, ws16);
  hipLaunchKernelGGL(attn_lstm_kernel, dim3(B_ / ROWS), dim3(NTHREADS), 0,
                     stream, X, M, curM, Wha, bha, Waa, baa, bhb, Wab, bab,
                     bih, bhh, Wp, bp, ws16, out);
}
